// LinformerSelfAttention_67851893342934
// MI455X (gfx1250) — matrix operand order repaired and checked by reading, not compiled, not run on hardware
//
#include <hip/hip_runtime.h>
#include <hip/hip_bf16.h>
#include <math.h>

// ---------------------------------------------------------------------------
// Linformer self-attention for MI455X (gfx1250, wave32, WMMA).
//
// Shapes: B=4, N=4096, D=1024, H=16, Dh=64, Kp=16.
// Dominant cost: 4 GEMMs [16384,1024]x[1024,1024] -> bf16 WMMA (f32 acc).
// Attention core (Kp=16) is bandwidth-trivial -> VALU kernel.
//
// Round-1 change: fp32->bf16 conversion now uses native casts so the backend
// emits hardware bf16 converts instead of a v_bfe/v_add3 RNE bit-twiddle
// chain on the staging critical path.
// ---------------------------------------------------------------------------

typedef __attribute__((ext_vector_type(16))) __bf16 v16bf;
typedef __attribute__((ext_vector_type(8)))  float  v8f;

// ---------------------------------------------------------------------------
// GEMM: Out[M,N] = X[M,K] @ W[N,K]^T + bias[N]     (torch Linear semantics)
// Block tile: 128(M) x 64(N), BK=32. 256 threads = 8 waves.
// Each wave computes one 16-row strip x 64 cols = 4 WMMA accumulators.
// ---------------------------------------------------------------------------
#define BM 128
#define BN 64
#define BKK 32
#define XS_STRIDE (BKK + 8)   // bf16 stride, padded vs bank conflicts
#define WS_STRIDE (BKK + 8)

__global__ __launch_bounds__(256) void gemm_xwt_bias_bf16(
    const float* __restrict__ X,     // [M,K]
    const float* __restrict__ W,     // [N,K]
    const float* __restrict__ bias,  // [N]
    float* __restrict__ Out,         // [M,N]
    int M, int N, int K)
{
    __shared__ __bf16 Xs[BM * XS_STRIDE];
    __shared__ __bf16 Ws[BN * WS_STRIDE];

    const int tid  = threadIdx.x;
    const int lane = tid & 31;
    const int wave = tid >> 5;                  // 0..7
    const int nBase = blockIdx.x * BN;
    const int mBase = blockIdx.y * BM;

    v8f acc[4];
    #pragma unroll
    for (int i = 0; i < 4; ++i)
        acc[i] = (v8f){0.f, 0.f, 0.f, 0.f, 0.f, 0.f, 0.f, 0.f};

    // staging assignments
    const int xRow  = tid >> 1;                 // 0..127
    const int xColH = (tid & 1) * 16;           // 0 / 16
    const int wRow  = tid >> 2;                 // 0..63
    const int wColQ = (tid & 3) * 8;            // 0,8,16,24

    const float* xSrc = X + (size_t)(mBase + xRow) * K + xColH;
    const float* wSrc = W + (size_t)(nBase + wRow) * K + wColQ;

    for (int k0 = 0; k0 < K; k0 += BKK) {
        // ---- stage X tile (128x32 fp32 -> bf16 in LDS) ----
        {
            const float* src = xSrc + k0;
            __bf16* dst = &Xs[xRow * XS_STRIDE + xColH];
            #pragma unroll
            for (int i = 0; i < 16; i += 4) {
                float4 f = *reinterpret_cast<const float4*>(src + i);
                dst[i + 0] = (__bf16)f.x; dst[i + 1] = (__bf16)f.y;
                dst[i + 2] = (__bf16)f.z; dst[i + 3] = (__bf16)f.w;
            }
        }
        // ---- stage W tile (64x32 fp32 -> bf16 in LDS, row n = out col) ----
        {
            const float* src = wSrc + k0;
            __bf16* dst = &Ws[wRow * WS_STRIDE + wColQ];
            #pragma unroll
            for (int i = 0; i < 8; i += 4) {
                float4 f = *reinterpret_cast<const float4*>(src + i);
                dst[i + 0] = (__bf16)f.x; dst[i + 1] = (__bf16)f.y;
                dst[i + 2] = (__bf16)f.z; dst[i + 3] = (__bf16)f.w;
            }
        }
        // prefetch next K tile into cache hierarchy (global_prefetch_b8)
        if (k0 + BKK < K) {
            __builtin_prefetch(xSrc + k0 + BKK, 0, 0);
            __builtin_prefetch(wSrc + k0 + BKK, 0, 0);
        }
        __syncthreads();

        // ---- A fragment: 16x32 bf16 per ISA layout ----
        // lane<16: row M=lane,   K in {0..7, 16..23}
        // lane>=16: row M=lane-16, K in {8..15, 24..31}
        v16bf a;
        {
            const int m  = wave * 16 + (lane & 15);
            const int kb = (lane < 16) ? 0 : 8;
            #pragma unroll
            for (int j = 0; j < 16; ++j) {
                const int kk = kb + j + ((j >> 3) << 3);
                a[j] = Xs[m * XS_STRIDE + kk];
            }
        }
        // ---- 4 B fragments + WMMA ----
        #pragma unroll
        for (int s = 0; s < 4; ++s) {
            v16bf bfm;
            const int n   = s * 16 + (lane & 15);
            const int kb2 = (lane < 16) ? 0 : 8;
            #pragma unroll
            for (int j = 0; j < 16; ++j)
                bfm[j] = Ws[n * WS_STRIDE + kb2 + j + ((j >> 3) << 3)];      // B[k][n] = W[n][k]
            acc[s] = __builtin_amdgcn_wmma_f32_16x16x32_bf16(
                false, a, false, bfm, (short)0, acc[s], false, false);
        }
        __syncthreads();
    }

    // ---- epilogue: C layout (VGPR r -> M = r or 8+r; col = lane&15) ----
    const int col16 = lane & 15;
    const int mOff  = (lane < 16) ? 0 : 8;
    #pragma unroll
    for (int s = 0; s < 4; ++s) {
        const int outCol = nBase + s * 16 + col16;
        const float bv = bias[outCol];
        #pragma unroll
        for (int r = 0; r < 8; ++r) {
            const int outRow = mBase + wave * 16 + mOff + r;
            Out[(size_t)outRow * N + outCol] = acc[s][r] + bv;
        }
    }
}

// ---------------------------------------------------------------------------
// Low-rank sequence compression: k_lin[b,h,kp,d] = sum_n E[h,kp,n]*k[b,n,h*64+d]
// One wave per (b,h, which). M=16(kp), N=64(d), K=4096(n). 4 accumulators.
// ---------------------------------------------------------------------------
#define ES_STRIDE (32 + 8)
#define KS_STRIDE (64 + 8)

__global__ __launch_bounds__(32) void seq_compress_bf16(
    const float* __restrict__ Kmat,  // [B,4096,1024]
    const float* __restrict__ Vmat,  // [B,4096,1024]
    const float* __restrict__ E,     // [16,16,4096]
    const float* __restrict__ F,     // [16,16,4096]
    float* __restrict__ Klin,        // [B,16,16,64]
    float* __restrict__ Vlin)        // [B,16,16,64]
{
    __shared__ __bf16 Es[16 * ES_STRIDE];
    __shared__ __bf16 Ks[32 * KS_STRIDE];

    const int bh  = blockIdx.x;          // 0..63
    const int b   = bh >> 4;
    const int h   = bh & 15;
    const int sel = blockIdx.y;          // 0 = K/E, 1 = V/F
    const int lane = threadIdx.x;        // 0..31

    const float* src  = sel ? Vmat : Kmat;
    const float* proj = sel ? F : E;
    float* dst        = sel ? Vlin : Klin;

    v8f acc[4];
    #pragma unroll
    for (int i = 0; i < 4; ++i)
        acc[i] = (v8f){0.f, 0.f, 0.f, 0.f, 0.f, 0.f, 0.f, 0.f};

    const int eRow  = lane & 15;              // kp row
    const int eColH = (lane >> 4) * 16;       // 0 / 16
    const float* eBase = proj + (size_t)h * 16 * 4096 + (size_t)eRow * 4096 + eColH;
    const float* kBase = src + (size_t)b * 4096 * 1024 + (size_t)h * 64;

    for (int n0 = 0; n0 < 4096; n0 += 32) {
        // stage E tile 16x32
        {
            const float* s0 = eBase + n0;
            __bf16* d0 = &Es[eRow * ES_STRIDE + eColH];
            #pragma unroll
            for (int i = 0; i < 16; i += 4) {
                float4 f = *reinterpret_cast<const float4*>(s0 + i);
                d0[i + 0] = (__bf16)f.x; d0[i + 1] = (__bf16)f.y;
                d0[i + 2] = (__bf16)f.z; d0[i + 3] = (__bf16)f.w;
            }
        }
        // stage K/V tile 32(n) x 64(d) : each lane loads one row of 64 floats
        {
            const float* s0 = kBase + (size_t)(n0 + lane) * 1024;
            __bf16* d0 = &Ks[lane * KS_STRIDE];
            #pragma unroll
            for (int i = 0; i < 64; i += 4) {
                float4 f = *reinterpret_cast<const float4*>(s0 + i);
                d0[i + 0] = (__bf16)f.x; d0[i + 1] = (__bf16)f.y;
                d0[i + 2] = (__bf16)f.z; d0[i + 3] = (__bf16)f.w;
            }
        }
        __syncthreads();

        v16bf a;
        {
            const int m  = lane & 15;
            const int kb = (lane < 16) ? 0 : 8;
            #pragma unroll
            for (int j = 0; j < 16; ++j) {
                const int kk = kb + j + ((j >> 3) << 3);
                a[j] = Es[m * ES_STRIDE + kk];
            }
        }
        #pragma unroll
        for (int s = 0; s < 4; ++s) {
            v16bf bfm;
            const int d   = s * 16 + (lane & 15);
            const int kb2 = (lane < 16) ? 0 : 8;
            #pragma unroll
            for (int j = 0; j < 16; ++j)
                bfm[j] = Ks[(kb2 + j + ((j >> 3) << 3)) * KS_STRIDE + d];
            acc[s] = __builtin_amdgcn_wmma_f32_16x16x32_bf16(
                false, a, false, bfm, (short)0, acc[s], false, false);
        }
        __syncthreads();
    }

    const int col16 = lane & 15;
    const int mOff  = (lane < 16) ? 0 : 8;
    #pragma unroll
    for (int s = 0; s < 4; ++s) {
        #pragma unroll
        for (int r = 0; r < 8; ++r) {
            const int kp = mOff + r;
            dst[((size_t)bh * 16 + kp) * 64 + s * 16 + col16] = acc[s][r];
        }
    }
}

// ---------------------------------------------------------------------------
// Attention core: scores[n,kp] = q[n]·k_lin[kp] / 8 ; softmax over kp (16);
// out[n] = sum_kp p[kp] * v_lin[kp]. One thread per query row.
// ---------------------------------------------------------------------------
__global__ __launch_bounds__(256) void attn_core(
    const float* __restrict__ Q,     // [B,4096,1024]
    const float* __restrict__ Klin,  // [B,16,16,64]
    const float* __restrict__ Vlin,  // [B,16,16,64]
    float* __restrict__ Outp)        // [B,4096,1024]
{
    __shared__ float kl[16 * 64];
    __shared__ float vl[16 * 64];

    const int bh = blockIdx.x;       // 0..63
    const int b  = bh >> 4;
    const int h  = bh & 15;
    const int t  = threadIdx.x;

    for (int i = t; i < 1024; i += 256) {
        kl[i] = Klin[(size_t)bh * 1024 + i];
        vl[i] = Vlin[(size_t)bh * 1024 + i];
    }
    __syncthreads();

    const int n = blockIdx.y * 256 + t;
    const float* qrow = Q + ((size_t)b * 4096 + n) * 1024 + h * 64;

    float sc[16];
    #pragma unroll
    for (int kp = 0; kp < 16; ++kp) sc[kp] = 0.f;

    #pragma unroll 4
    for (int d = 0; d < 64; ++d) {
        const float qd = qrow[d];
        #pragma unroll
        for (int kp = 0; kp < 16; ++kp)
            sc[kp] += qd * kl[kp * 64 + d];
    }

    float mx = sc[0] * 0.125f;
    #pragma unroll
    for (int kp = 0; kp < 16; ++kp) {
        sc[kp] *= 0.125f;
        mx = fmaxf(mx, sc[kp]);
    }
    float sum = 0.f;
    #pragma unroll
    for (int kp = 0; kp < 16; ++kp) {
        sc[kp] = __expf(sc[kp] - mx);
        sum += sc[kp];
    }
    const float inv = 1.0f / sum;
    #pragma unroll
    for (int kp = 0; kp < 16; ++kp) sc[kp] *= inv;

    float* orow = Outp + ((size_t)b * 4096 + n) * 1024 + h * 64;
    #pragma unroll 4
    for (int d = 0; d < 64; ++d) {
        float o = 0.f;
        #pragma unroll
        for (int kp = 0; kp < 16; ++kp)
            o += sc[kp] * vl[kp * 64 + d];
        orow[d] = o;
    }
}

// ---------------------------------------------------------------------------
// Launch
// ---------------------------------------------------------------------------
extern "C" void kernel_launch(void* const* d_in, const int* in_sizes, int n_in,
                              void* d_out, int out_size, void* d_ws, size_t ws_size,
                              hipStream_t stream) {
    (void)in_sizes; (void)n_in; (void)out_size; (void)ws_size;

    const float* x  = (const float*)d_in[0];
    const float* Wq = (const float*)d_in[1];
    const float* bq = (const float*)d_in[2];
    const float* Wk = (const float*)d_in[3];
    const float* bk = (const float*)d_in[4];
    const float* Wv = (const float*)d_in[5];
    const float* bv = (const float*)d_in[6];
    const float* Wo = (const float*)d_in[7];
    const float* bo = (const float*)d_in[8];
    const float* E  = (const float*)d_in[9];
    const float* F  = (const float*)d_in[10];
    float* out = (float*)d_out;

    const int Mrows = 4 * 4096;      // B * SEQ
    const int Dm    = 1024;

    float* ws = (float*)d_ws;
    const size_t big = (size_t)Mrows * Dm;   // 16,777,216 floats
    float* q    = ws;
    float* kbuf = ws + big;
    float* vbuf = ws + 2 * big;
    float* attn = ws + 3 * big;
    float* klin = ws + 4 * big;              // 4*16*16*64 = 65536 floats
    float* vlin = klin + 65536;

    const dim3 gemmGrid(Dm / BN, Mrows / BM);   // (16, 128)
    const dim3 gemmBlk(256);

    // Q, K, V projections
    gemm_xwt_bias_bf16<<<gemmGrid, gemmBlk, 0, stream>>>(x, Wq, bq, q,    Mrows, Dm, Dm);
    gemm_xwt_bias_bf16<<<gemmGrid, gemmBlk, 0, stream>>>(x, Wk, bk, kbuf, Mrows, Dm, Dm);
    gemm_xwt_bias_bf16<<<gemmGrid, gemmBlk, 0, stream>>>(x, Wv, bv, vbuf, Mrows, Dm, Dm);

    // k_lin = E @ K, v_lin = F @ V
    seq_compress_bf16<<<dim3(64, 2), dim3(32), 0, stream>>>(kbuf, vbuf, E, F, klin, vlin);

    // softmax attention over Kp=16
    attn_core<<<dim3(64, 16), dim3(256), 0, stream>>>(q, klin, vlin, attn);

    // output projection
    gemm_xwt_bias_bf16<<<gemmGrid, gemmBlk, 0, stream>>>(attn, Wo, bo, out, Mrows, Dm, Dm);
}
